// DGCNN_58737972740763
// MI455X (gfx1250) — matrix operand order repaired and checked by reading, not compiled, not run on hardware
//
#include <hip/hip_runtime.h>
#include <hip/hip_bf16.h>

typedef __attribute__((ext_vector_type(2)))  float  v2f;
typedef __attribute__((ext_vector_type(8)))  float  v8f;
typedef __attribute__((ext_vector_type(16))) __bf16 v16bf;

#define NNODES 102400
#define NGRAPH 256
#define PPER   400
#define NEDGE  (NNODES * 32)
#define TOPK   30
#define CCH    97

#if defined(__has_builtin)
#if __has_builtin(__builtin_amdgcn_wmma_f32_16x16x4_f32)
#define HAS_WMMA_F32 1
#endif
#endif

// ---------------------------------------------------------------------------
// degree / normalization
// ---------------------------------------------------------------------------
__global__ void k_init_deg(float* __restrict__ deg, int n) {
  int i = blockIdx.x * blockDim.x + threadIdx.x;
  if (i < n) deg[i] = 1.0f;                     // self-loop weight baseline
}

__global__ void k_deg(const int* __restrict__ ei, int E, float* __restrict__ deg) {
  long idx = (long)blockIdx.x * blockDim.x + threadIdx.x;
  if (idx >= E) return;
  int s = ei[idx], d = ei[(long)E + idx];
  if (s != d) atomicAdd(&deg[d], 1.0f);
}

__global__ void k_dis(const float* __restrict__ deg, float* __restrict__ dis, int n) {
  int i = blockIdx.x * blockDim.x + threadIdx.x;
  if (i < n) dis[i] = rsqrtf(deg[i]);
}

// ---------------------------------------------------------------------------
// WMMA GEMM: out[N,32] = A[N,K] @ W[K,32]   (lda = row stride of A)
// One wave computes one 16x16 output tile; block of 4 waves covers 32x32.
// ---------------------------------------------------------------------------
__global__ void k_gemm32(const float* __restrict__ A, int lda, int K,
                         const float* __restrict__ W, float* __restrict__ out) {
  const int wave = threadIdx.x >> 5;            // 0..3
  const int lane = threadIdx.x & 31;
  const int m0   = blockIdx.x * 32 + (wave >> 1) * 16;
  const int n0   = (wave & 1) * 16;
  const int half = lane >> 4;                   // 0 or 1
  const int l    = lane & 15;
  const int mrow = m0 + l;
  const int n    = n0 + l;
  v8f c = {};
#ifdef HAS_WMMA_F32
  for (int k0 = 0; k0 < K; k0 += 4) {
    const int ka = k0 + half * 2;
    v2f a; a[0] = A[(long)mrow * lda + ka]; a[1] = A[(long)mrow * lda + ka + 1];
    v2f b; b[0] = W[ka * 32 + n];           b[1] = W[(ka + 1) * 32 + n];
    c = __builtin_amdgcn_wmma_f32_16x16x4_f32(false, a, false, b,
                                              (short)0, c, false, false);
  }
#else
  for (int k0 = 0; k0 < K; k0 += 32) {
    v16bf a, b;
    for (int i = 0; i < 4; ++i) {
      int k = k0 + half * 16 + 2 * i;
      a[2*i]   = (__bf16)A[(long)mrow * lda + k];
      a[2*i+1] = (__bf16)A[(long)mrow * lda + k + 1];
    }
    for (int i = 4; i < 8; ++i) {
      int k = k0 + half * 16 + 2 * i;
      a[2*i]   = (__bf16)A[(long)mrow * lda + k];
      a[2*i+1] = (__bf16)A[(long)mrow * lda + k + 1];
    }
    for (int j = 0; j < 8; ++j) {
      int k = k0 + half * 16 + 2 * j;
      b[2*j]   = (__bf16)W[k * 32 + n];
      b[2*j+1] = (__bf16)W[(k + 1) * 32 + n];
    }
    c = __builtin_amdgcn_wmma_f32_16x16x32_bf16(false, a, false, b,
                                                (short)0, c, false, false);
  }
#endif
  for (int r = 0; r < 8; ++r)
    out[(long)(m0 + half * 8 + r) * 32 + n0 + l] = c[r];
}

// ---------------------------------------------------------------------------
// GCN aggregation (32-channel layers)
// ---------------------------------------------------------------------------
__global__ void k_initacc(const float* __restrict__ hw, const float* __restrict__ dis,
                          const float* __restrict__ b, float* __restrict__ acc, int n) {
  long idx = (long)blockIdx.x * blockDim.x + threadIdx.x;
  if (idx >= (long)n * 32) return;
  int node = (int)(idx >> 5), ch = (int)idx & 31;
  float d = dis[node];
  acc[idx] = b[ch] + d * d * hw[idx];
}

__global__ void k_edge32(const int* __restrict__ ei, int E,
                         const float* __restrict__ dis,
                         const float* __restrict__ hw, float* __restrict__ acc) {
  long idx = (long)blockIdx.x * blockDim.x + threadIdx.x;
  if (idx >= (long)E * 8) return;
  int e  = (int)(idx >> 3);
  int c0 = ((int)idx & 7) * 4;
  int s = ei[e], d = ei[(long)E + e];
  if (s == d) return;
  float w = dis[s] * dis[d];
  const float* hs = hw + (long)s * 32 + c0;
  float*       ad = acc + (long)d * 32 + c0;
  atomicAdd(ad + 0, w * hs[0]);
  atomicAdd(ad + 1, w * hs[1]);
  atomicAdd(ad + 2, w * hs[2]);
  atomicAdd(ad + 3, w * hs[3]);
}

__global__ void k_tanh32(const float* __restrict__ acc, float* __restrict__ h,
                         int off, int n) {
  long idx = (long)blockIdx.x * blockDim.x + threadIdx.x;
  if (idx >= (long)n * 32) return;
  int node = (int)(idx >> 5), ch = (int)idx & 31;
  h[(long)node * CCH + off + ch] = tanhf(acc[idx]);
}

// ---------------------------------------------------------------------------
// layer 4 (out width 1)
// ---------------------------------------------------------------------------
__global__ void k_gemv4(const float* __restrict__ h, const float* __restrict__ W4,
                        float* __restrict__ hw1, int n) {
  int i = blockIdx.x * blockDim.x + threadIdx.x;
  if (i >= n) return;
  const float* row = h + (long)i * CCH + 64;
  float acc = 0.f;
  for (int j = 0; j < 32; ++j) acc += row[j] * W4[j];
  hw1[i] = acc;
}

__global__ void k_initacc1(const float* __restrict__ hw1, const float* __restrict__ dis,
                           const float* __restrict__ b4, float* __restrict__ acc1, int n) {
  int i = blockIdx.x * blockDim.x + threadIdx.x;
  if (i >= n) return;
  float d = dis[i];
  acc1[i] = b4[0] + d * d * hw1[i];
}

__global__ void k_edge1(const int* __restrict__ ei, int E,
                        const float* __restrict__ dis,
                        const float* __restrict__ hw1, float* __restrict__ acc1) {
  long idx = (long)blockIdx.x * blockDim.x + threadIdx.x;
  if (idx >= E) return;
  int s = ei[idx], d = ei[(long)E + idx];
  if (s == d) return;
  atomicAdd(&acc1[d], dis[s] * dis[d] * hw1[s]);
}

__global__ void k_tanh1(const float* __restrict__ acc1, float* __restrict__ h, int n) {
  int i = blockIdx.x * blockDim.x + threadIdx.x;
  if (i >= n) return;
  h[(long)i * CCH + 96] = tanhf(acc1[i]);
}

// ---------------------------------------------------------------------------
// sort-pool: one block per graph; rank-by-counting with stable tie-break
// ---------------------------------------------------------------------------
__global__ void k_sortpool(const float* __restrict__ h, float* __restrict__ seq) {
  __shared__ float keys[PPER];
  __shared__ int   sel[TOPK];
  const int g   = blockIdx.x;
  const int tid = threadIdx.x;
  const long base = (long)g * PPER;
  for (int j = tid; j < PPER; j += blockDim.x)
    keys[j] = h[(base + j) * CCH + 96];
  __syncthreads();
  for (int j = tid; j < PPER; j += blockDim.x) {
    float kj = keys[j];
    int rank = 0;
    for (int i = 0; i < PPER; ++i) {
      float ki = keys[i];
      rank += (ki > kj) || (ki == kj && i < j);   // stable desc sort
    }
    if (rank < TOPK) sel[rank] = j;
  }
  __syncthreads();
  float* sq = seq + (long)g * (TOPK * CCH);
  for (int t = tid; t < TOPK * CCH; t += blockDim.x) {
    int k = t / CCH, c = t - k * CCH;
    sq[t] = h[(base + sel[k]) * CCH + c];
  }
}

// ---------------------------------------------------------------------------
// CNN + MLP head: one block (128 threads) per graph, all stages in LDS
// ---------------------------------------------------------------------------
__global__ void k_head(const float* __restrict__ seq,
                       const float* __restrict__ cw5, const float* __restrict__ cb5,
                       const float* __restrict__ cw6, const float* __restrict__ cb6,
                       const float* __restrict__ fw1, const float* __restrict__ fb1,
                       const float* __restrict__ fw2, const float* __restrict__ fb2,
                       float* __restrict__ out) {
  __shared__ float s_out5[16 * 30];
  __shared__ float s_p[16 * 15];
  __shared__ float s_flat[352];
  __shared__ float s_h1[128];
  __shared__ float red[128];
  const int g   = blockIdx.x;
  const int tid = threadIdx.x;
  const float* sq = seq + (long)g * (TOPK * CCH);

  // conv5: stride-97 kernel-97 => per-node dot over 97 channels
  for (int idx = tid; idx < 16 * 30; idx += blockDim.x) {
    int o = idx / 30, t = idx - o * 30;
    float acc = cb5[o];
    const float* w  = cw5 + o * CCH;
    const float* sp = sq + t * CCH;
    for (int c = 0; c < CCH; ++c) acc += w[c] * sp[c];
    s_out5[o * 30 + t] = fmaxf(acc, 0.f);
  }
  __syncthreads();
  // maxpool(2,2)
  for (int idx = tid; idx < 16 * 15; idx += blockDim.x) {
    int o = idx / 15, j = idx - o * 15;
    s_p[o * 15 + j] = fmaxf(s_out5[o * 30 + 2 * j], s_out5[o * 30 + 2 * j + 1]);
  }
  __syncthreads();
  // conv6: 16ch -> 32ch, kernel 5
  for (int idx = tid; idx < 32 * 11; idx += blockDim.x) {
    int o = idx / 11, t = idx - o * 11;
    float acc = cb6[o];
    for (int c = 0; c < 16; ++c) {
      const float* w = cw6 + (o * 16 + c) * 5;
      const float* p = s_p + c * 15 + t;
      for (int k = 0; k < 5; ++k) acc += w[k] * p[k];
    }
    s_flat[o * 11 + t] = fmaxf(acc, 0.f);
  }
  __syncthreads();
  // fc1: 352 -> 128, relu
  {
    float acc = fb1[tid];
    for (int j = 0; j < 352; ++j) acc += s_flat[j] * fw1[j * 128 + tid];
    s_h1[tid] = fmaxf(acc, 0.f);
  }
  __syncthreads();
  // fc2: 128 -> 1, tree reduction
  red[tid] = s_h1[tid] * fw2[tid];
  __syncthreads();
  for (int stride = 64; stride > 0; stride >>= 1) {
    if (tid < stride) red[tid] += red[tid + stride];
    __syncthreads();
  }
  if (tid == 0) out[g] = red[0] + fb2[0];
}

// ---------------------------------------------------------------------------
// launcher
// ---------------------------------------------------------------------------
extern "C" void kernel_launch(void* const* d_in, const int* in_sizes, int n_in,
                              void* d_out, int out_size, void* d_ws, size_t ws_size,
                              hipStream_t stream) {
  const float* x   = (const float*)d_in[0];
  const int*   ei  = (const int*)d_in[1];
  // d_in[2] = batch (implicit: node / PPER)
  const float* W1 = (const float*)d_in[3],  *b1 = (const float*)d_in[4];
  const float* W2 = (const float*)d_in[5],  *b2 = (const float*)d_in[6];
  const float* W3 = (const float*)d_in[7],  *b3 = (const float*)d_in[8];
  const float* W4 = (const float*)d_in[9],  *b4 = (const float*)d_in[10];
  const float* cw5 = (const float*)d_in[11], *cb5 = (const float*)d_in[12];
  const float* cw6 = (const float*)d_in[13], *cb6 = (const float*)d_in[14];
  const float* fw1 = (const float*)d_in[15], *fb1 = (const float*)d_in[16];
  const float* fw2 = (const float*)d_in[17], *fb2 = (const float*)d_in[18];
  float* out = (float*)d_out;

  const int N = NNODES, E = NEDGE;
  float* f    = (float*)d_ws;
  float* dis  = f;                 // N
  float* deg  = f + (long)N;       // N
  float* hw   = f + (long)2  * N;  // 32N
  float* acc  = f + (long)34 * N;  // 32N
  float* hw1  = f + (long)66 * N;  // N
  float* acc1 = f + (long)67 * N;  // N
  float* h    = f + (long)68 * N;  // 97N
  float* seq  = f + (long)165 * N; // G*2910

  const int TB = 256;
  const int gN    = (N + TB - 1) / TB;
  const int gE    = (E + TB - 1) / TB;
  const int gN32  = (N * 32 + TB - 1) / TB;
  const int gE8   = (int)(((long)E * 8 + TB - 1) / TB);
  const int gGemm = N / 32;

  // degree + normalization
  k_init_deg<<<gN, TB, 0, stream>>>(deg, N);
  k_deg<<<gE, TB, 0, stream>>>(ei, E, deg);
  k_dis<<<gN, TB, 0, stream>>>(deg, dis, N);

  // layer 1: x[N,128] @ W1[128,32]
  k_gemm32<<<gGemm, 128, 0, stream>>>(x, 128, 128, W1, hw);
  k_initacc<<<gN32, TB, 0, stream>>>(hw, dis, b1, acc, N);
  k_edge32<<<gE8, TB, 0, stream>>>(ei, E, dis, hw, acc);
  k_tanh32<<<gN32, TB, 0, stream>>>(acc, h, 0, N);

  // layer 2: h[:,0:32] @ W2
  k_gemm32<<<gGemm, 128, 0, stream>>>(h, CCH, 32, W2, hw);
  k_initacc<<<gN32, TB, 0, stream>>>(hw, dis, b2, acc, N);
  k_edge32<<<gE8, TB, 0, stream>>>(ei, E, dis, hw, acc);
  k_tanh32<<<gN32, TB, 0, stream>>>(acc, h, 32, N);

  // layer 3: h[:,32:64] @ W3
  k_gemm32<<<gGemm, 128, 0, stream>>>(h + 32, CCH, 32, W3, hw);
  k_initacc<<<gN32, TB, 0, stream>>>(hw, dis, b3, acc, N);
  k_edge32<<<gE8, TB, 0, stream>>>(ei, E, dis, hw, acc);
  k_tanh32<<<gN32, TB, 0, stream>>>(acc, h, 64, N);

  // layer 4: h[:,64:96] @ W4[32,1]
  k_gemv4<<<gN, TB, 0, stream>>>(h, W4, hw1, N);
  k_initacc1<<<gN, TB, 0, stream>>>(hw1, dis, b4, acc1, N);
  k_edge1<<<gE, TB, 0, stream>>>(ei, E, dis, hw1, acc1);
  k_tanh1<<<gN, TB, 0, stream>>>(acc1, h, N);

  // sort-pool + head
  k_sortpool<<<NGRAPH, 256, 0, stream>>>(h, seq);
  k_head<<<NGRAPH, 128, 0, stream>>>(seq, cw5, cb5, cw6, cb6,
                                     fw1, fb1, fw2, fb2, out);
}
